// L2MultiheadAttention_30030411334458
// MI455X (gfx1250) — matrix operand order repaired and checked by reading, not compiled, not run on hardware
//
#include <hip/hip_runtime.h>

// ---------------------------------------------------------------------------
// L2 multi-head attention, algebraically collapsed:
//   M_h   = scale * W_h^T V_h                     (16 heads of 64x64)
//   vtld  = q @ M_h                               (q == k, same projection)
//   O     = softmax_s(2*scale*q.k - scale*kk[s]) @ vtld   (flash-style)
//   out   = O @ out_w^T + out_b
// All big GEMMs use v_wmma_f32_16x16x32_bf16 (fp32 accum), softmax in fp32.
// K-loops are software-pipelined (ping-pong tiles) so WMMAs never wait on
// loads issued in the same stage. T=1024, B=4, H=16, d=64, D=1024.
// ---------------------------------------------------------------------------

typedef __bf16 bf16_t;
typedef __attribute__((ext_vector_type(16))) __bf16 v16bf;
typedef __attribute__((ext_vector_type(8)))  __bf16 v8bf;
typedef __attribute__((ext_vector_type(8)))  float  v8f;

#define TT   1024
#define NB   4
#define NH   16
#define HD   64
#define DD   1024
#define TBR  4096   // T*NB rows

static __device__ __forceinline__ v8f wmma_bf16(v16bf a, v16bf b, v8f c) {
  return __builtin_amdgcn_wmma_f32_16x16x32_bf16(false, a, false, b, (short)0, c,
                                                 false, false);
}

// A-tile (16xK32, row-major source, rowbase = tile origin).
// Order used here (same in ld_bt): lane m=lane&15, hi=lane>>4; K = hi*16 + 2v (+1) -> contiguous.
static __device__ __forceinline__ v16bf ld_a(const bf16_t* rowbase, int ld,
                                             int lane, int k0) {
  const int m = lane & 15, hi = lane >> 4;
  const bf16_t* p = rowbase + m * ld + k0 + hi * 16;
  v8bf lo = *(const v8bf*)(p);
  v8bf hh = *(const v8bf*)(p + 8);
  v16bf a;
#pragma unroll
  for (int i = 0; i < 8; ++i) { a[i] = lo[i]; a[i + 8] = hh[i]; }
  return a;
}

// B-tile (K32x16) read from B^T stored row-major (n-major, contiguous k).
// Order used here (same in ld_a): lane n=lane&15, hi=lane>>4; K = hi*16 + 2v (+1) -> contiguous.
static __device__ __forceinline__ v16bf ld_bt(const bf16_t* colbaseT, int ld,
                                              int lane, int k0) {
  const int n = lane & 15, hi = lane >> 4;
  return *(const v16bf*)(colbaseT + n * ld + k0 + hi * 16);
}

// ---------------- converts / transposes -----------------------------------

__global__ void k_convert_bf16(const float* __restrict__ src,
                               bf16_t* __restrict__ dst, int n) {
  int i = blockIdx.x * blockDim.x + threadIdx.x;
  int stride = gridDim.x * blockDim.x;
  for (; i < n; i += stride) dst[i] = (bf16_t)src[i];
}

// dst[n*1024 + k] = src[k*1024 + n]   (1024x1024, fp32 -> bf16)
__global__ void k_transpose1024_bf16(const float* __restrict__ src,
                                     bf16_t* __restrict__ dst) {
  __shared__ float tile[32][33];
  int x = blockIdx.x * 32 + threadIdx.x;
#pragma unroll
  for (int i = 0; i < 32; i += 8)
    tile[threadIdx.y + i][threadIdx.x] = src[(blockIdx.y * 32 + threadIdx.y + i) * DD + x];
  __syncthreads();
  int xo = blockIdx.y * 32 + threadIdx.x;
#pragma unroll
  for (int i = 0; i < 32; i += 8)
    dst[(blockIdx.x * 32 + threadIdx.y + i) * DD + xo] =
        (bf16_t)tile[threadIdx.x][threadIdx.y + i];
}

// ---------------- QK projection: q = x @ Wq, scatter to [b][h][t][d] -------
// 32x64 per wave, ping-pong K pipeline.

__global__ __launch_bounds__(128) void k_gemm_qproj(
    const bf16_t* __restrict__ xbf, const bf16_t* __restrict__ qwT,
    bf16_t* __restrict__ qbf) {
  const int w = threadIdx.x >> 5, lane = threadIdx.x & 31;
  const int n = lane & 15, hi = lane >> 4;
  const int m0 = blockIdx.y * 128 + w * 32;
  const int n0 = blockIdx.x * 64;
  v8f acc[2][4] = {};

  auto ldA = [&](v16bf* a, int k0) {
    a[0] = ld_a(xbf + (size_t)m0 * DD, DD, lane, k0);
    a[1] = ld_a(xbf + (size_t)(m0 + 16) * DD, DD, lane, k0);
  };
  auto ldB = [&](v16bf* b, int k0) {
#pragma unroll
    for (int j = 0; j < 4; ++j)
      b[j] = ld_bt(qwT + (size_t)(n0 + 16 * j) * DD, DD, lane, k0);
  };
  auto mm = [&](const v16bf* a, const v16bf* b) {
#pragma unroll
    for (int mi = 0; mi < 2; ++mi)
#pragma unroll
      for (int j = 0; j < 4; ++j)
        acc[mi][j] = wmma_bf16(a[mi], b[j], acc[mi][j]);
  };

  v16bf aA[2], aB[2], bA[4], bB[4];
  ldA(aA, 0); ldB(bA, 0);
  for (int k0 = 0; k0 < DD; k0 += 64) {
    ldA(aB, k0 + 32); ldB(bB, k0 + 32);
    mm(aA, bA);
    int kn = (k0 + 64) & (DD - 1);     // wrap: dummy prefetch on last iter
    ldA(aA, kn); ldB(bA, kn);
    mm(aB, bB);
  }
#pragma unroll
  for (int mi = 0; mi < 2; ++mi)
#pragma unroll
    for (int j = 0; j < 4; ++j)
#pragma unroll
      for (int r = 0; r < 8; ++r) {
        int row = m0 + 16 * mi + r + 8 * hi;   // row = t*4 + b
        int col = n0 + 16 * j + n;             // col = h*64 + d
        int t = row >> 2, bb = row & 3, h = col >> 6, dd = col & 63;
        qbf[((size_t)(bb * NH + h) * TT + t) * HD + dd] = (bf16_t)acc[mi][j][r];
      }
}

// ---------------- kk[b,h,t] = sum_d q^2 ------------------------------------

__global__ void k_kk(const bf16_t* __restrict__ qbf, float* __restrict__ kk) {
  int idx = blockIdx.x * blockDim.x + threadIdx.x;   // NB*NH*TT threads
  const bf16_t* p = qbf + (size_t)idx * HD;
  float s = 0.f;
#pragma unroll
  for (int d = 0; d < HD; ++d) { float v = (float)p[d]; s += v * v; }
  kk[idx] = s;
}

// ---------------- M_h^T = (scale * W_h^T V_h)^T  (bf16, n-major) -----------

__global__ __launch_bounds__(128) void k_gemm_headM(
    const bf16_t* __restrict__ qwT, const bf16_t* __restrict__ vwT,
    bf16_t* __restrict__ MhT) {
  const int h = blockIdx.x;
  const int w = threadIdx.x >> 5, lane = threadIdx.x & 31;
  const int n = lane & 15, hi = lane >> 4;
  const int i0 = w * 16;
  v8f acc[4] = {};

  auto ldA = [&](v16bf* a, int k0) {
    a[0] = ld_a(qwT + (size_t)(h * HD + i0) * DD, DD, lane, k0);
  };
  auto ldB = [&](v16bf* b, int k0) {
#pragma unroll
    for (int j = 0; j < 4; ++j)
      b[j] = ld_bt(vwT + (size_t)(h * HD + 16 * j) * DD, DD, lane, k0);
  };
  auto mm = [&](const v16bf* a, const v16bf* b) {
#pragma unroll
    for (int j = 0; j < 4; ++j) acc[j] = wmma_bf16(a[0], b[j], acc[j]);
  };

  v16bf aA[1], aB[1], bA[4], bB[4];
  ldA(aA, 0); ldB(bA, 0);
  for (int k0 = 0; k0 < DD; k0 += 64) {
    ldA(aB, k0 + 32); ldB(bB, k0 + 32);
    mm(aA, bA);
    int kn = (k0 + 64) & (DD - 1);
    ldA(aA, kn); ldB(bA, kn);
    mm(aB, bB);
  }
#pragma unroll
  for (int j = 0; j < 4; ++j)
#pragma unroll
    for (int r = 0; r < 8; ++r)
      MhT[(size_t)(h * HD + 16 * j + n) * HD + i0 + r + 8 * hi] =
          (bf16_t)(0.125f * acc[j][r]);
}

// ---------------- vtld^T[b][h][d][t] = (q @ M_h)^T -------------------------
// K=64: load everything, then 8 WMMAs behind one wait.

__global__ __launch_bounds__(128) void k_gemm_vtilde(
    const bf16_t* __restrict__ qbf, const bf16_t* __restrict__ MhT,
    bf16_t* __restrict__ vtT) {
  const int w = threadIdx.x >> 5, lane = threadIdx.x & 31;
  const int n = lane & 15, hi = lane >> 4;
  const int bh = blockIdx.y, h = bh & 15;
  const int t0 = blockIdx.x * 64 + w * 16;
  const bf16_t* qs = qbf + (size_t)bh * TT * HD;
  const bf16_t* mh = MhT + (size_t)h * HD * HD;

  v16bf a0 = ld_a(qs + (size_t)t0 * HD, HD, lane, 0);
  v16bf a1 = ld_a(qs + (size_t)t0 * HD, HD, lane, 32);
  v16bf b[4][2];
#pragma unroll
  for (int j = 0; j < 4; ++j) {
    b[j][0] = ld_bt(mh + (size_t)(16 * j) * HD, HD, lane, 0);
    b[j][1] = ld_bt(mh + (size_t)(16 * j) * HD, HD, lane, 32);
  }
  v8f acc[4] = {};
#pragma unroll
  for (int j = 0; j < 4; ++j) {
    acc[j] = wmma_bf16(a0, b[j][0], acc[j]);
    acc[j] = wmma_bf16(a1, b[j][1], acc[j]);
  }
  bf16_t* vs = vtT + (size_t)bh * HD * TT;
#pragma unroll
  for (int j = 0; j < 4; ++j)
#pragma unroll
    for (int r = 0; r < 8; ++r)
      vs[(size_t)(16 * j + n) * TT + t0 + r + 8 * hi] = (bf16_t)acc[j][r];
}

// ---------------- fused flash attention ------------------------------------
// logits[t,s] = 2*scale*(q_t . q_s) - scale*kk[s]   (qq[t] cancels in softmax)
// K-tiles prefetched one 32-block ahead; V-tiles issued before the score
// WMMAs and consumed after the softmax VALU burst (latency hidden).

__global__ __launch_bounds__(128) void k_attn(
    const bf16_t* __restrict__ qbf, const bf16_t* __restrict__ vtT,
    const float* __restrict__ kkbuf, bf16_t* __restrict__ pxav) {
  const float scale = 0.125f;
  const int w = threadIdx.x >> 5, lane = threadIdx.x & 31;
  const int n = lane & 15, hi = lane >> 4, m = lane & 15;
  const int bh = blockIdx.y, bb = bh >> 4, h = bh & 15;
  const int t0 = blockIdx.x * 64 + w * 16;
  const bf16_t* qs = qbf + (size_t)bh * TT * HD;
  const bf16_t* vs = vtT + (size_t)bh * HD * TT;
  const float* kks = kkbuf + (size_t)bh * TT;

  __shared__ __align__(32) __bf16 pls[4][16][32];   // wave-private P staging

  const v16bf aq0 = ld_a(qs + (size_t)t0 * HD, HD, lane, 0);
  const v16bf aq1 = ld_a(qs + (size_t)t0 * HD, HD, lane, 32);

  float mrow[8], lrow[8];
  v8f o[4] = {};
#pragma unroll
  for (int r = 0; r < 8; ++r) { mrow[r] = -3.0e38f; lrow[r] = 0.f; }

  auto load_k = [&](v16bf* bk, int s) {
    bk[0] = ld_bt(qs + (size_t)s * HD, HD, lane, 0);
    bk[1] = ld_bt(qs + (size_t)s * HD, HD, lane, 32);
    bk[2] = ld_bt(qs + (size_t)(s + 16) * HD, HD, lane, 0);
    bk[3] = ld_bt(qs + (size_t)(s + 16) * HD, HD, lane, 32);
  };

  auto attn_block = [&](int s0, const v16bf* bk) {
    // prefetch V tiles + kk bias early; consumed only after softmax
    v16bf bv[4];
#pragma unroll
    for (int j = 0; j < 4; ++j)
      bv[j] = ld_bt(vs + (size_t)(16 * j) * TT, TT, lane, s0);
    const float kb0 = kks[s0 + n];
    const float kb1 = kks[s0 + 16 + n];

    v8f c0 = {}, c1 = {};
    c0 = wmma_bf16(aq0, bk[0], c0);
    c0 = wmma_bf16(aq1, bk[1], c0);
    c1 = wmma_bf16(aq0, bk[2], c1);
    c1 = wmma_bf16(aq1, bk[3], c1);

    float p0[8], p1[8];
#pragma unroll
    for (int r = 0; r < 8; ++r) {
      float sc0 = 2.f * scale * c0[r] - scale * kb0;
      float sc1 = 2.f * scale * c1[r] - scale * kb1;
      float v = fmaxf(sc0, sc1);
      v = fmaxf(v, __shfl_xor(v, 1, 32));
      v = fmaxf(v, __shfl_xor(v, 2, 32));
      v = fmaxf(v, __shfl_xor(v, 4, 32));
      v = fmaxf(v, __shfl_xor(v, 8, 32));
      float mn = fmaxf(mrow[r], v);
      float f = __expf(mrow[r] - mn);
      mrow[r] = mn;
      float e0 = __expf(sc0 - mn);
      float e1 = __expf(sc1 - mn);
      p0[r] = e0; p1[r] = e1;
      float rs = e0 + e1;
      rs += __shfl_xor(rs, 1, 32);
      rs += __shfl_xor(rs, 2, 32);
      rs += __shfl_xor(rs, 4, 32);
      rs += __shfl_xor(rs, 8, 32);
      lrow[r] = lrow[r] * f + rs;
#pragma unroll
      for (int j = 0; j < 4; ++j) o[j][r] *= f;
    }
    // C-layout -> row-major P in wave-private LDS (same-wave DS is in-order)
#pragma unroll
    for (int r = 0; r < 8; ++r) {
      pls[w][r + 8 * hi][n]      = (__bf16)p0[r];
      pls[w][r + 8 * hi][16 + n] = (__bf16)p1[r];
    }
    const __bf16* prow = &pls[w][m][0];
    v8bf plo = *(const v8bf*)(prow + hi * 16);
    v8bf phi = *(const v8bf*)(prow + hi * 16 + 8);
    v16bf pa;
#pragma unroll
    for (int i = 0; i < 8; ++i) { pa[i] = plo[i]; pa[i + 8] = phi[i]; }
#pragma unroll
    for (int j = 0; j < 4; ++j) o[j] = wmma_bf16(pa, bv[j], o[j]);
  };

  v16bf bkA[4], bkB[4];
  load_k(bkA, 0);
  for (int s0 = 0; s0 < TT; s0 += 64) {
    load_k(bkB, s0 + 32);
    attn_block(s0, bkA);
    load_k(bkA, (s0 + 64) & (TT - 1));   // wrap: dummy prefetch on last iter
    attn_block(s0 + 32, bkB);
  }

#pragma unroll
  for (int j = 0; j < 4; ++j)
#pragma unroll
    for (int r = 0; r < 8; ++r) {
      int t = t0 + r + 8 * hi;
      int col = h * HD + 16 * j + n;
      float val = o[j][r] / lrow[r];
      pxav[(size_t)(t * NB + bb) * DD + col] = (bf16_t)val;
    }
}

// ---------------- output projection: out = O @ out_w^T + out_b -------------

__global__ __launch_bounds__(128) void k_gemm_outproj(
    const bf16_t* __restrict__ pxav, const bf16_t* __restrict__ owbf,
    const float* __restrict__ bias, float* __restrict__ out) {
  const int w = threadIdx.x >> 5, lane = threadIdx.x & 31;
  const int n = lane & 15, hi = lane >> 4;
  const int m0 = blockIdx.y * 128 + w * 32;
  const int n0 = blockIdx.x * 64;
  v8f acc[2][4] = {};

  auto ldA = [&](v16bf* a, int k0) {
    a[0] = ld_a(pxav + (size_t)m0 * DD, DD, lane, k0);
    a[1] = ld_a(pxav + (size_t)(m0 + 16) * DD, DD, lane, k0);
  };
  auto ldB = [&](v16bf* b, int k0) {
#pragma unroll
    for (int j = 0; j < 4; ++j)   // out_w row j is exactly B^T row j
      b[j] = ld_bt(owbf + (size_t)(n0 + 16 * j) * DD, DD, lane, k0);
  };
  auto mm = [&](const v16bf* a, const v16bf* b) {
#pragma unroll
    for (int mi = 0; mi < 2; ++mi)
#pragma unroll
      for (int j = 0; j < 4; ++j)
        acc[mi][j] = wmma_bf16(a[mi], b[j], acc[mi][j]);
  };

  v16bf aA[2], aB[2], bA[4], bB[4];
  ldA(aA, 0); ldB(bA, 0);
  for (int k0 = 0; k0 < DD; k0 += 64) {
    ldA(aB, k0 + 32); ldB(bB, k0 + 32);
    mm(aA, bA);
    int kn = (k0 + 64) & (DD - 1);
    ldA(aA, kn); ldB(bA, kn);
    mm(aB, bB);
  }
#pragma unroll
  for (int mi = 0; mi < 2; ++mi)
#pragma unroll
    for (int j = 0; j < 4; ++j)
#pragma unroll
      for (int r = 0; r < 8; ++r) {
        int row = m0 + 16 * mi + r + 8 * hi;
        int col = n0 + 16 * j + n;
        out[(size_t)row * DD + col] = acc[mi][j][r] + bias[col];
      }
}

// ---------------------------------------------------------------------------

extern "C" void kernel_launch(void* const* d_in, const int* in_sizes, int n_in,
                              void* d_out, int out_size, void* d_ws, size_t ws_size,
                              hipStream_t stream) {
  const float* x  = (const float*)d_in[0];
  const float* qw = (const float*)d_in[1];
  const float* vw = (const float*)d_in[2];
  const float* ow = (const float*)d_in[3];
  const float* ob = (const float*)d_in[4];
  float* out = (float*)d_out;

  char* ws = (char*)d_ws;
  size_t off = 0;
  auto alloc = [&](size_t bytes) {
    void* p = ws + off;
    off += (bytes + 255) & ~(size_t)255;
    return p;
  };
  bf16_t* xbf  = (bf16_t*)alloc((size_t)TBR * DD * 2);      // x, bf16
  bf16_t* qwT  = (bf16_t*)alloc((size_t)DD * DD * 2);       // Wq^T (n-major)
  bf16_t* vwT  = (bf16_t*)alloc((size_t)DD * DD * 2);       // Wv^T (n-major)
  bf16_t* owbf = (bf16_t*)alloc((size_t)DD * DD * 2);       // out_w, bf16
  bf16_t* qbf  = (bf16_t*)alloc((size_t)NB * NH * TT * HD * 2); // q [b][h][t][d]
  float*  kkb  = (float*)alloc((size_t)NB * NH * TT * 4);   // kk [b][h][t]
  bf16_t* MhT  = (bf16_t*)alloc((size_t)NH * HD * HD * 2);  // M_h^T
  bf16_t* vtT  = (bf16_t*)alloc((size_t)NB * NH * HD * TT * 2); // vtld^T [b][h][d][t]
  bf16_t* pxav = (bf16_t*)alloc((size_t)TBR * DD * 2);      // attention output
  (void)ws_size; (void)in_sizes; (void)n_in; (void)out_size;

  k_convert_bf16<<<1024, 256, 0, stream>>>(x, xbf, TBR * DD);
  k_convert_bf16<<<1024, 256, 0, stream>>>(ow, owbf, DD * DD);
  k_transpose1024_bf16<<<dim3(32, 32), dim3(32, 8), 0, stream>>>(qw, qwT);
  k_transpose1024_bf16<<<dim3(32, 32), dim3(32, 8), 0, stream>>>(vw, vwT);

  k_gemm_qproj<<<dim3(16, 32), 128, 0, stream>>>(xbf, qwT, qbf);
  k_kk<<<dim3(256), 256, 0, stream>>>(qbf, kkb);
  k_gemm_headM<<<dim3(16), 128, 0, stream>>>(qwT, vwT, MhT);
  k_gemm_vtilde<<<dim3(16, 64), 128, 0, stream>>>(qbf, MhT, vtT);
  k_attn<<<dim3(16, 64), 128, 0, stream>>>(qbf, vtT, kkb, pxav);
  k_gemm_outproj<<<dim3(16, 32), 128, 0, stream>>>(pxav, owbf, ob, out);
}
